// RelPosEmb1D_7267084664878
// MI455X (gfx1250) — matrix operand order repaired and checked by reading, not compiled, not run on hardware
//
#include <hip/hip_runtime.h>

typedef __attribute__((ext_vector_type(16))) __bf16 v16bf;
typedef __attribute__((ext_vector_type(8)))  __bf16 v8bf;
typedef __attribute__((ext_vector_type(8)))  float  v8f;
typedef __attribute__((ext_vector_type(4)))  float  v4f;

#define TOK  2048
#define DH   64
#define BH   32                 // b*h planes (fixed by reference: 2*16)
#define NREL (2 * TOK - 1)
#define RELN (NREL * DH)
#define QN   (BH * TOK * DH)

// Preconverted bf16 hi/lo operand images (device .bss, no workspace needed).
__device__ __bf16 g_relhi[RELN];
__device__ __bf16 g_rello[RELN];
__device__ __bf16 g_qhi[QN];
__device__ __bf16 g_qlo[QN];

// ---------------------------------------------------------------------------
// Pre-passes: split fp32 -> bf16 hi + bf16 lo residual ONCE, so the hot GEMM
// loop has zero conversion VALU. (rel: 1MB, q: 16MB; ~1.5us of HBM traffic.)
// ---------------------------------------------------------------------------
__global__ void split_rel_k(const float* __restrict__ rel) {
  const int i = blockIdx.x * 256 + threadIdx.x;
  if (i < RELN) {
    const float v = rel[i];
    const __bf16 h = (__bf16)v;
    g_relhi[i] = h;
    g_rello[i] = (__bf16)(v - (float)h);
  }
}

__global__ void split_q_k(const float* __restrict__ q) {
  const int i = blockIdx.x * 256 + threadIdx.x;
  if (i < QN) {
    const float v = q[i];
    const __bf16 h = (__bf16)v;
    g_qhi[i] = h;
    g_qlo[i] = (__bf16)(v - (float)h);
  }
}

// ---------------------------------------------------------------------------
// out[z,i,j] = sum_d q[z,i,d] * rel[j-i+L-1, d]
// One wave = one 16-row x 128-col output strip. Sliding window of 9 rel
// tiles (16x16) serves 8 output tiles; skew extraction via a LINEAR per-wave
// 16x144 LDS strip (no modulo, DS ops pair into 2addr forms).
// fp32-class accuracy from bf16 split: ah*bh + ah*bl + al*bh.
// ---------------------------------------------------------------------------
__global__ __launch_bounds__(128)
void relpos_skew_gemm(float* __restrict__ out) {
  __shared__ float lds[4][16][144];  // 4 waves x 16 rows x 9 tile slots (36KB)

  const int lane = threadIdx.x & 31;
  const int wv   = threadIdx.x >> 5;
  const int n    = lane & 15;        // N col (B) / M row (A) / out row (epi)
  const int hb   = lane >> 4;        // lane half
  const int z    = blockIdx.z;                       // b*h plane
  const int i0   = blockIdx.y * 16;                  // output row block
  const int jbase = (blockIdx.x * 4 + wv) * 128;     // wave's column strip
  const int r_strip = jbase - i0 + (TOK - 1) - 16;   // rel-row origin of strip

  // ---- A operands, preconverted. A k order used here (contiguous): lane half hb holds
  // K run [16hb, 16hb+16): two 16B loads merged per chunk.
  v16bf ah[2], al[2];
  const size_t arow = ((size_t)z * TOK + i0 + n) * DH;
#pragma unroll
  for (int kc = 0; kc < 2; ++kc) {
    const size_t a0 = arow + kc * 32 + hb * 16;
    const v8bf h0 = *(const v8bf*)(g_qhi + a0);
    const v8bf h1 = *(const v8bf*)(g_qhi + a0 + 8);
    const v8bf l0 = *(const v8bf*)(g_qlo + a0);
    const v8bf l1 = *(const v8bf*)(g_qlo + a0 + 8);
    ah[kc] = __builtin_shufflevector(h0, h1, 0, 1, 2, 3, 4, 5, 6, 7,
                                     8, 9, 10, 11, 12, 13, 14, 15);
    al[kc] = __builtin_shufflevector(l0, l1, 0, 1, 2, 3, 4, 5, 6, 7,
                                     8, 9, 10, 11, 12, 13, 14, 15);
  }

  float* outplane = out + (size_t)z * TOK * TOK;

  for (int g = 0; g <= 8; ++g) {
    // ---- rel tile g: T[m,c] = sum_d q[i0+m,d] * rel[r_strip+16g+c, d]
    int rr = r_strip + 16 * g + n;                 // this lane's rel row (B col)
    rr = rr < 0 ? 0 : (rr > NREL - 1 ? NREL - 1 : rr);
    const size_t brow = (size_t)rr * DH;

    v8f acc = {};
#pragma unroll
    for (int kc = 0; kc < 2; ++kc) {
      // B k order used here (same as A): lane half hb holds K = 16hb..16hb+15 (contig 32B run)
      const size_t boff = brow + kc * 32 + hb * 16;
      const v16bf bh = *(const v16bf*)(g_relhi + boff);
      const v16bf bl = *(const v16bf*)(g_rello + boff);
      acc = __builtin_amdgcn_wmma_f32_16x16x32_bf16(false, ah[kc], false, bh,
                                                    (short)0, acc, false, false);
      acc = __builtin_amdgcn_wmma_f32_16x16x32_bf16(false, ah[kc], false, bl,
                                                    (short)0, acc, false, false);
      acc = __builtin_amdgcn_wmma_f32_16x16x32_bf16(false, al[kc], false, bh,
                                                    (short)0, acc, false, false);
    }

    // ---- stash tile g at linear cols [16g, 16g+16) (C/D WMMA layout)
    const int colw = 16 * g + n;
#pragma unroll
    for (int v = 0; v < 8; ++v)
      lds[wv][v + 8 * hb][colw] = acc[v];

    if (g == 0) continue;

    // per-wave LDS RAW: drain DS ops, then skew-extract output tile jt = g-1.
    asm volatile("s_wait_dscnt 0" ::: "memory");
    const int jt  = g - 1;
    const int ch0 = 8 * hb;                        // lane: 8 contig cols of row n
    const int off0 = 16 * g + ch0 - n;             // run start, 1..143 (no wrap)
    float vals[8];
#pragma unroll
    for (int c = 0; c < 8; ++c)                    // 8 consecutive LDS floats
      vals[c] = lds[wv][n][off0 + c];
    const v4f s0 = {vals[0], vals[1], vals[2], vals[3]};
    const v4f s1 = {vals[4], vals[5], vals[6], vals[7]};
    float* dst = outplane + (size_t)(i0 + n) * TOK + (jbase + 16 * jt + ch0);
    __builtin_nontemporal_store(s0, (v4f*)dst);    // stream 512MB past L2
    __builtin_nontemporal_store(s1, (v4f*)(dst + 4));
  }
}

extern "C" void kernel_launch(void* const* d_in, const int* in_sizes, int n_in,
                              void* d_out, int out_size, void* d_ws, size_t ws_size,
                              hipStream_t stream) {
  (void)n_in; (void)out_size; (void)d_ws; (void)ws_size;
  const float* q   = (const float*)d_in[0];   // [b,h,2048,64] f32
  const float* rel = (const float*)d_in[1];   // [4095,64] f32
  float* out = (float*)d_out;                 // [b,h,2048,2048] f32

  int planes = in_sizes[0] / (TOK * DH);      // b*h = 32
  if (planes > BH) planes = BH;

  split_rel_k<<<(RELN + 255) / 256, 256, 0, stream>>>(rel);
  split_q_k<<<(QN + 255) / 256, 256, 0, stream>>>(q);

  dim3 grid(TOK / 512, TOK / 16, planes);     // (4, 128, 32)
  relpos_skew_gemm<<<grid, 128, 0, stream>>>(out);
}
